// DiscreteVAE_84567906058598
// MI455X (gfx1250) — matrix operand order repaired and checked by reading, not compiled, not run on hardware
//
#include <hip/hip_runtime.h>
#include <stdint.h>

// DiscreteVAE (Point-BERT dVAE) forward for MI455X (gfx1250, wave32, WMMA).
// GEMMs: v_wmma_f32_16x16x32_f16, 8 waves/block on a 32x64 macro tile.
// A/B slabs are double-buffered in LDS and staged with
// GLOBAL_LOAD_ASYNC_TO_LDS_B128 (ASYNCcnt): each lane issues exactly 3 async
// copies per 64-deep K step, so `s_wait_asynccnt 3` overlaps the next tile's
// fetch with the current tile's two WMMAs. All f16 weights (~50MB) stay
// L2-resident (192MB). Global BatchNorm / GroupNorm are two-pass via f32
// atomics in the GEMM epilogue.

typedef _Float16 half_t;
typedef __attribute__((ext_vector_type(16))) _Float16 v16h;
typedef __attribute__((ext_vector_type(8)))  _Float16 v8h;
typedef __attribute__((ext_vector_type(8)))  float    v8f;

namespace {
constexpr int kNB   = 2;
constexpr int kN    = 8192;   // points per batch
constexpr int kGRP  = 512;    // groups per batch
constexpr int kS    = 128;    // points per group
constexpr int kMG   = kNB * kGRP;   // 1024 groups total
constexpr int kRows = kMG * kS;     // 131072 per-point rows
constexpr int kNT   = 8192;   // vocab
constexpr float kEps = 1e-5f;
}

// ---------------------------------------------------------------- utilities

__global__ void zero_k(float* p, int n) {
  int i = blockIdx.x * 256 + threadIdx.x;
  if (i < n) p[i] = 0.f;
}

// f32 weight [Co,K] -> f16 [Co,K64] zero padded
__global__ void conv_pad_k(const float* __restrict__ src, int K, int K64,
                           half_t* __restrict__ dst) {
  int o = blockIdx.x;
  for (int c = threadIdx.x; c < K64; c += 256)
    dst[(size_t)o * K64 + c] = (half_t)((c < K) ? src[(size_t)o * K + c] : 0.f);
}

// codebook [8192,256] -> f16 transpose [256,8192] (weight layout [Co,Ci])
__global__ void conv_cbT_k(const float* __restrict__ cb, half_t* __restrict__ dst) {
  int n = blockIdx.x;                      // 0..255
  for (int k = threadIdx.x; k < 8192; k += 256)
    dst[(size_t)n * 8192 + k] = (half_t)cb[(size_t)k * 256 + n];
}

// ---------------------------------------------------------------- WMMA GEMM
// C[M,N] = act( A[M,K64]f16 @ Bw[N,K64]^T f16 + bias[N] )
// smode: 0 none, 1 BatchNorm stats per channel, 2 GroupNorm(4) stats per (b,cg)
// Block: 256 threads = 8 waves in 2(M) x 4(N) grid; macro tile 32x64.
// K step 64: stage 32x64 A (4KB) + 64x64 B (8KB) per buffer, double buffered.
__global__ void wmma_gemm_k(const half_t* __restrict__ A,
                            const half_t* __restrict__ Bw,
                            const float* __restrict__ bias,
                            int M, int N, int K64,
                            half_t* __restrict__ o16, float* __restrict__ o32,
                            int act, float* __restrict__ stats,
                            int smode, int rpb) {
  __shared__ __align__(64) half_t smA[2][32 * 64];   // 8KB
  __shared__ __align__(64) half_t smB[2][64 * 64];   // 16KB
  const int t = threadIdx.x;              // 0..255
  const int wave = t >> 5, l = t & 31;
  const int wm = wave >> 2, wn = wave & 3;        // 2 x 4 wave grid
  const int h = l >> 4, lo = l & 15;
  const int m0 = blockIdx.y * 32;
  const int n0 = blockIdx.x * 64;

  // staging: 16B chunks (8 halves); per lane: 1 A chunk + 2 B chunks
  const int cr = t >> 3, cc = (t & 7) * 8;        // cr 0..31, cc 0..56
  const size_t aoff  = (size_t)(m0 + cr) * K64 + cc;      // M multiple of 32
  const int    bro0  = n0 + cr,      bro1 = n0 + cr + 32;
  const size_t boff0 = (size_t)((bro0 < N) ? bro0 : 0) * K64 + cc;
  const size_t boff1 = (size_t)((bro1 < N) ? bro1 : 0) * K64 + cc;
  uint32_t ldsA[2], ldsB0[2], ldsB1[2];
  for (int bf = 0; bf < 2; ++bf) {
    ldsA[bf]  = (uint32_t)(uintptr_t)&smA[bf][cr * 64 + cc];
    ldsB0[bf] = (uint32_t)(uintptr_t)&smB[bf][cr * 64 + cc];
    ldsB1[bf] = (uint32_t)(uintptr_t)&smB[bf][(cr + 32) * 64 + cc];
  }

  auto stage = [&](int bf, int kt) {
    asm volatile("global_load_async_to_lds_b128 %0, %1, off"
                 :: "v"(ldsA[bf]), "v"((uint64_t)(uintptr_t)(A + aoff + kt))
                 : "memory");
    asm volatile("global_load_async_to_lds_b128 %0, %1, off"
                 :: "v"(ldsB0[bf]), "v"((uint64_t)(uintptr_t)(Bw + boff0 + kt))
                 : "memory");
    asm volatile("global_load_async_to_lds_b128 %0, %1, off"
                 :: "v"(ldsB1[bf]), "v"((uint64_t)(uintptr_t)(Bw + boff1 + kt))
                 : "memory");
  };

  v8f acc = {};
  const int nIter = K64 >> 6;
  stage(0, 0);
  for (int it = 0; it < nIter; ++it) {
    const int cur = it & 1;
    const bool hasNext = (it + 1) < nIter;
    if (hasNext) {
      stage(cur ^ 1, (it + 1) << 6);
      __builtin_prefetch(Bw + boff0 + ((it + 2) << 6), 0, 1);  // L2 stream hint
      // 3 copies just issued; wait until the 3 older (current buffer) retire
      asm volatile("s_wait_asynccnt 0x3" ::: "memory");
    } else {
      asm volatile("s_wait_asynccnt 0x0" ::: "memory");
    }
    __syncthreads();
    // two 16x16x32 WMMAs over the 64-deep slab
    const half_t* sa = &smA[cur][(wm * 16 + lo) * 64];
    const half_t* sb = &smB[cur][(wn * 16 + lo) * 64];
#pragma unroll
    for (int sub = 0; sub < 64; sub += 32) {
      union { v16h v; v8h p[2]; } a;
      a.p[0] = *(const v8h*)(sa + sub + 8 * h);         // K {8h..8h+7}
      a.p[1] = *(const v8h*)(sa + sub + 16 + 8 * h);    // K {16+8h..23+8h}
      union { v16h v; v8h p[2]; } b;
      b.p[0] = *(const v8h*)(sb + sub + 8 * h);         // K {8h..8h+7}
      b.p[1] = *(const v8h*)(sb + sub + 16 + 8 * h);    // K {16+8h..23+8h}
      acc = __builtin_amdgcn_wmma_f32_16x16x32_f16(
          false, a.v, false, b.v, (short)0, acc, false, false);
    }
    __syncthreads();          // readers done before this buffer is restaged
  }

  const int bn = n0 + wn * 16 + lo;
  const float bval = (bias && bn < N) ? bias[bn] : 0.f;
  for (int v = 0; v < 8; ++v) {
    int mrow = m0 + wm * 16 + v + 8 * h;
    if (mrow >= M || bn >= N) continue;
    float y = acc[v] + bval;
    if (stats && smode) {
      int g = (smode == 1) ? bn : ((mrow / rpb) * 4 + bn / (N / 4));
      atomicAdd(&stats[g], y);
      atomicAdd(&stats[(smode == 1 ? N : 8) + g], y * y);
    }
    if (act == 1)      y = fmaxf(y, 0.f);
    else if (act == 2) y = (y >= 0.f) ? y : 0.2f * y;
    if (o16) o16[(size_t)mrow * N + bn] = (half_t)y;
    if (o32) o32[(size_t)mrow * N + bn] = y;
  }
}

// ------------------------------------------------------- norm (BN/GN) apply
__global__ void norm_apply_k(const half_t* __restrict__ raw, int N,
                             const float* __restrict__ stats, int mode, int rpb,
                             float cnt, const float* __restrict__ gamma,
                             const float* __restrict__ beta, int act,
                             const half_t* __restrict__ addp,
                             half_t* __restrict__ o16, float* __restrict__ o32) {
  size_t r = blockIdx.x;
  for (int n = threadIdx.x; n < N; n += 256) {
    int g = (mode == 1) ? n : (int)(r / (size_t)rpb) * 4 + n / (N / 4);
    float mu  = stats[g] / cnt;
    float var = stats[(mode == 1 ? N : 8) + g] / cnt - mu * mu;
    float y = ((float)raw[r * N + n] - mu) * rsqrtf(var + kEps) * gamma[n] + beta[n];
    if (act == 1)      y = fmaxf(y, 0.f);
    else if (act == 2) y = (y >= 0.f) ? y : 0.2f * y;
    if (addp) y += (float)addp[r * N + n];
    if (o16) o16[r * N + n] = (half_t)y;
    if (o32) o32[r * N + n] = y;
  }
}

// ---------------------------------------------------------- max over axis k
__global__ void max_over_k(const half_t* __restrict__ in, int k, int C,
                           half_t* __restrict__ out1,
                           half_t* __restrict__ out2, int stride2) {
  size_t r = blockIdx.x;
  for (int c = threadIdx.x; c < C; c += 256) {
    float m = -3.4e38f;
    for (int j = 0; j < k; ++j)
      m = fmaxf(m, (float)in[(r * k + j) * (size_t)C + c]);
    out1[r * (size_t)C + c] = (half_t)m;
    if (out2) out2[r * (size_t)stride2 + c] = (half_t)m;
  }
}

// -------------------------------------------- kNN(128 of 8192) + gathering
__global__ void knn_gather_k(const float* __restrict__ xyz,
                             const float* __restrict__ emb,
                             const int* __restrict__ cidx,
                             float* __restrict__ centers,
                             half_t* __restrict__ neighA,   // [131072,64] padded
                             half_t* __restrict__ embA) {   // [131072,128]
  __shared__ float dist[kN];      // 32KB of the 320KB WGP LDS
  __shared__ float rv[256];
  __shared__ int   ri[256];
  __shared__ int   sel[kS];
  const int bg = blockIdx.x, b = bg / kGRP, t = threadIdx.x;
  const int ci = cidx[bg];
  const float cx = xyz[((size_t)b * kN + ci) * 3 + 0];
  const float cy = xyz[((size_t)b * kN + ci) * 3 + 1];
  const float cz = xyz[((size_t)b * kN + ci) * 3 + 2];
  for (int p = t; p < kN; p += 256) {
    float dx = xyz[((size_t)b * kN + p) * 3 + 0] - cx;
    float dy = xyz[((size_t)b * kN + p) * 3 + 1] - cy;
    float dz = xyz[((size_t)b * kN + p) * 3 + 2] - cz;
    dist[p] = dx * dx + dy * dy + dz * dz;
  }
  __syncthreads();
  for (int s = 0; s < kS; ++s) {           // 128 iterative argmins
    float bv = 3.4e38f; int bi = 0x7fffffff;
    for (int p = t; p < kN; p += 256) {
      float v = dist[p];
      if (v < bv || (v == bv && p < bi)) { bv = v; bi = p; }
    }
    rv[t] = bv; ri[t] = bi; __syncthreads();
    for (int st = 128; st > 0; st >>= 1) {
      if (t < st) {
        float v2 = rv[t + st]; int i2 = ri[t + st];
        if (v2 < rv[t] || (v2 == rv[t] && i2 < ri[t])) { rv[t] = v2; ri[t] = i2; }
      }
      __syncthreads();
    }
    if (t == 0) { sel[s] = ri[0]; dist[ri[0]] = 3.4e38f; }
    __syncthreads();
  }
  if (t == 0) {
    centers[bg * 3 + 0] = cx; centers[bg * 3 + 1] = cy; centers[bg * 3 + 2] = cz;
  }
  for (int s = t; s < kS; s += 256) {
    int p = sel[s];
    size_t r = (size_t)bg * kS + s;
    neighA[r * 64 + 0] = (half_t)(xyz[((size_t)b * kN + p) * 3 + 0] - cx);
    neighA[r * 64 + 1] = (half_t)(xyz[((size_t)b * kN + p) * 3 + 1] - cy);
    neighA[r * 64 + 2] = (half_t)(xyz[((size_t)b * kN + p) * 3 + 2] - cz);
    for (int c = 3; c < 64; ++c) neighA[r * 64 + c] = (half_t)0.f;
  }
  for (int e = t; e < kS * 128; e += 256) {
    int s = e >> 7, c = e & 127;
    int p = sel[s];
    embA[((size_t)bg * kS + s) * 128 + c] =
        (half_t)emb[((size_t)b * kN + p) * 128 + c];
  }
}

// kNN(16 of 512) on center coords, per batch
__global__ void knn16_k(const float* __restrict__ centers, int* __restrict__ idx16) {
  __shared__ float dist[kGRP];
  __shared__ float rv[128];
  __shared__ int   ri[128];
  const int bg = blockIdx.x, b = bg >> 9, t = threadIdx.x;   // 128 threads
  const float cx = centers[bg * 3 + 0], cy = centers[bg * 3 + 1], cz = centers[bg * 3 + 2];
  for (int p = t; p < kGRP; p += 128) {
    float dx = centers[((size_t)b * kGRP + p) * 3 + 0] - cx;
    float dy = centers[((size_t)b * kGRP + p) * 3 + 1] - cy;
    float dz = centers[((size_t)b * kGRP + p) * 3 + 2] - cz;
    dist[p] = dx * dx + dy * dy + dz * dz;
  }
  __syncthreads();
  for (int s = 0; s < 16; ++s) {
    float bv = 3.4e38f; int bi = 0x7fffffff;
    for (int p = t; p < kGRP; p += 128) {
      float v = dist[p];
      if (v < bv || (v == bv && p < bi)) { bv = v; bi = p; }
    }
    rv[t] = bv; ri[t] = bi; __syncthreads();
    for (int st = 64; st > 0; st >>= 1) {
      if (t < st) {
        float v2 = rv[t + st]; int i2 = ri[t + st];
        if (v2 < rv[t] || (v2 == rv[t] && i2 < ri[t])) { rv[t] = v2; ri[t] = i2; }
      }
      __syncthreads();
    }
    if (t == 0) { idx16[bg * 16 + s] = ri[0]; dist[ri[0]] = 3.4e38f; }
    __syncthreads();
  }
}

// graph_feature: gf[bg,j,:] = [ f[nbr]-f[bg] , f[bg] ]   (2C wide)
__global__ void graph_feat_k(const half_t* __restrict__ f,
                             const int* __restrict__ idx16, int C,
                             half_t* __restrict__ gf) {
  __shared__ half_t fq[1024];
  const int bg = blockIdx.x, b = bg >> 9, t = threadIdx.x;
  for (int c = t; c < C; c += 256) fq[c] = f[(size_t)bg * C + c];
  __syncthreads();
  for (int j = 0; j < 16; ++j) {
    int nb = idx16[bg * 16 + j];
    const half_t* fn = f + (size_t)(b * kGRP + nb) * C;
    half_t* out = gf + ((size_t)bg * 16 + j) * (size_t)(2 * C);
    for (int c = t; c < C; c += 256) {
      out[c]     = (half_t)((float)fn[c] - (float)fq[c]);
      out[C + c] = fq[c];
    }
  }
}

// encoder concat: [ broadcast(max f) | f ]  -> 512 wide
__global__ void enc_concat_k(const half_t* __restrict__ fg,
                             const half_t* __restrict__ f,
                             half_t* __restrict__ out) {
  size_t r = blockIdx.x; int m = (int)(r >> 7);
  for (int c = threadIdx.x; c < 512; c += 128)
    out[r * 512 + c] = (c < 256) ? fg[(size_t)m * 256 + c] : f[r * 256 + (c - 256)];
}

// gumbel softmax (tau=1), f32 logits -> f16 probs
__global__ void gumbel_softmax_k(const float* __restrict__ logits,
                                 const float* __restrict__ noise,
                                 half_t* __restrict__ soft) {
  __shared__ float red[256];
  const size_t r = blockIdx.x; const int t = threadIdx.x;
  const float* lr = logits + r * kNT;
  const float* nr = noise + r * kNT;
  float mx = -3.4e38f;
  for (int c = t; c < kNT; c += 256) mx = fmaxf(mx, lr[c] + nr[c]);
  red[t] = mx; __syncthreads();
  for (int st = 128; st > 0; st >>= 1) {
    if (t < st) red[t] = fmaxf(red[t], red[t + st]);
    __syncthreads();
  }
  mx = red[0]; __syncthreads();
  float sm = 0.f;
  for (int c = t; c < kNT; c += 256) sm += __expf(lr[c] + nr[c] - mx);
  red[t] = sm; __syncthreads();
  for (int st = 128; st > 0; st >>= 1) {
    if (t < st) red[t] += red[t + st];
    __syncthreads();
  }
  const float inv = 1.f / red[0];
  for (int c = t; c < kNT; c += 256)
    soft[r * kNT + c] = (half_t)(__expf(lr[c] + nr[c] - mx) * inv);
}

// decoder fine input: [ feature(256) | seed(2) | pf(3) | pad->320 ]
__global__ void dec_feat_k(const half_t* __restrict__ feat,
                           const float* __restrict__ coarse,
                           half_t* __restrict__ A) {
  size_t r = blockIdx.x;                 // 0..131071
  int m = (int)(r >> 7), j = (int)(r & 127);
  const float sx = (j & 1) ? 0.05f : -0.05f;
  const float sy = (j & 2) ? 0.05f : -0.05f;
  half_t* row = A + r * 320;
  for (int c = threadIdx.x; c < 320; c += 128) {
    float v;
    if (c < 256)       v = (float)feat[(size_t)m * 256 + c];
    else if (c == 256) v = sx;
    else if (c == 257) v = sy;
    else if (c < 261)  v = coarse[((size_t)m * 32 + (j >> 2)) * 3 + (c - 258)];
    else               v = 0.f;
    row[c] = (half_t)v;
  }
}

// fine = xyz_head + pf
__global__ void fine_add_k(const float* __restrict__ xyzraw,
                           const float* __restrict__ coarse,
                           float* __restrict__ fine) {
  int e = blockIdx.x * 256 + threadIdx.x;
  if (e >= kRows * 3) return;
  int c = e % 3, r = e / 3, m = r >> 7, j = r & 127;
  fine[e] = xyzraw[e] + coarse[((size_t)m * 32 + (j >> 2)) * 3 + c];
}

// ---------------------------------------------------------------- launch

extern "C" void kernel_launch(void* const* d_in, const int* in_sizes, int n_in,
                              void* d_out, int out_size, void* d_ws, size_t ws_size,
                              hipStream_t stream) {
  (void)in_sizes; (void)n_in; (void)out_size; (void)ws_size;

  // input indices: dict order, params flattened in sorted-key DFS order
  constexpr int I_XYZ = 0, I_EMB = 1, I_CIDX = 2, I_NOISE = 3;
  constexpr int I_CB = 4, I_DEC = 5, I_DG1 = 23, I_DG2 = 40, I_ENC = 57;
  auto F = [&](int i) { return (const float*)d_in[i]; };

  // ---- workspace bump allocator (deterministic; identical every call)
  uintptr_t base = (uintptr_t)d_ws; size_t off = 0;
  auto alloc = [&](size_t bytes) -> void* {
    uintptr_t p = base + off; off = (off + bytes + 255) & ~(size_t)255; return (void*)p;
  };
  auto ah = [&](size_t n) { return (half_t*)alloc(n * sizeof(half_t)); };
  auto af = [&](size_t n) { return (float*)alloc(n * sizeof(float)); };
  auto ai = [&](size_t n) { return (int*)alloc(n * sizeof(int)); };

  auto convW = [&](int idx, int Co, int K) -> half_t* {
    int K64 = (K + 63) & ~63;
    half_t* d = ah((size_t)Co * K64);
    conv_pad_k<<<dim3(Co), dim3(256), 0, stream>>>(F(idx), K, K64, d);
    return d;
  };
  auto gemm = [&](const half_t* A, const half_t* Bw, const float* bias,
                  int M, int N, int K64, half_t* o16, float* o32, int act,
                  float* stats, int smode, int rpb) {
    wmma_gemm_k<<<dim3((N + 63) / 64, M / 32), dim3(256), 0, stream>>>(
        A, Bw, bias, M, N, K64, o16, o32, act, stats, smode, rpb);
  };
  auto norm = [&](const half_t* raw, size_t R, int N, float* stats, int mode,
                  int rpb, float cnt, const float* g, const float* b, int act,
                  const half_t* addp, half_t* o16, float* o32) {
    norm_apply_k<<<dim3((unsigned)R), dim3(256), 0, stream>>>(
        raw, N, stats, mode, rpb, cnt, g, b, act, addp, o16, o32);
  };

  float* stats = af(2048);
  auto zstat = [&]() { zero_k<<<dim3(8), dim3(256), 0, stream>>>(stats, 2048); };

  // ---- output regions (coarse | fine | fine_emb | logits)
  float* out = (float*)d_out;
  float* o_coarse = out;
  float* o_fine   = out + 98304;
  float* o_femb   = out + 98304 + 393216;
  float* o_logits = out + 98304 + 393216 + 16777216;

  // ---- weights -> f16 (idempotent per call)
  half_t* wf_fw = convW(I_ENC + 9, 128, 3);        // K64=64
  half_t* wf_sw = convW(I_ENC + 11, 256, 128);
  half_t* wf_c1 = convW(I_ENC + 3, 512, 512);
  half_t* wf_c2 = convW(I_ENC + 5, 256, 512);
  half_t* cbT = ah((size_t)256 * 8192);
  conv_cbT_k<<<dim3(256), dim3(256), 0, stream>>>(F(I_CB), cbT);
  half_t* wm1  = convW(I_DEC + 11, 1024, 256);
  half_t* wm2  = convW(I_DEC + 13, 1024, 1024);
  half_t* wm3  = convW(I_DEC + 15, 96, 1024);
  half_t* wfc1 = convW(I_DEC + 5, 512, 261);       // K64=320
  half_t* wfc2 = convW(I_DEC + 9, 512, 512);
  half_t* wxyz = convW(I_DEC + 17, 3, 512);
  half_t* wemb = convW(I_DEC + 1, 128, 512);

  // ---- activations
  float*  centers = af((size_t)kMG * 3);
  half_t* neighA  = ah((size_t)kRows * 64);
  half_t* embA    = ah((size_t)kRows * 128);
  half_t* f128raw = ah((size_t)kRows * 128);
  half_t* f128n   = ah((size_t)kRows * 128);
  half_t* f256    = ah((size_t)kRows * 256);
  half_t* fg256   = ah((size_t)kMG * 256);
  half_t* cat512  = ah((size_t)kRows * 512);   // reused for fc2 raw
  half_t* c1raw   = ah((size_t)kRows * 512);   // reused for fc1 raw
  half_t* c1n     = ah((size_t)kRows * 512);   // reused for fc1 normed
  half_t* fc2n    = ah((size_t)kRows * 512);
  half_t* c2out   = ah((size_t)kRows * 256);
  half_t* logits0 = ah((size_t)kMG * 256);
  int*    idx16   = ai((size_t)kMG * 16);
  half_t* ftr     = ah((size_t)kMG * 1024);
  half_t* gf      = ah((size_t)kMG * 16 * 2048);
  half_t* yraw    = ah((size_t)kMG * 16 * 1024);
  half_t* yn      = ah((size_t)kMG * 16 * 1024);
  half_t* feats   = ah((size_t)kMG * 2304);
  half_t* l5raw   = ah((size_t)kMG * 8192);
  half_t* soft    = ah((size_t)kMG * 8192);
  half_t* sampled = ah((size_t)kMG * 256);
  half_t* feature = ah((size_t)kMG * 256);
  half_t* h1      = ah((size_t)kMG * 1024);
  half_t* h2      = ah((size_t)kMG * 1024);
  half_t* featA   = ah((size_t)kRows * 320);
  float*  xyzraw  = af((size_t)kRows * 3);

  // ================= stage A: grouping =================
  knn_gather_k<<<dim3(kMG), dim3(256), 0, stream>>>(
      F(I_XYZ), F(I_EMB), (const int*)d_in[I_CIDX], centers, neighA, embA);
  knn16_k<<<dim3(kMG), dim3(128), 0, stream>>>(centers, idx16);

  // ================= stage B: encoder =================
  zstat();
  gemm(neighA, wf_fw, F(I_ENC + 6), kRows, 128, 64, f128raw, nullptr, 0, stats, 1, 0);
  norm(f128raw, kRows, 128, stats, 1, 0, (float)kRows, F(I_ENC + 8), F(I_ENC + 7),
       1, embA, f128n, nullptr);                           // relu(bn(.)) + emb
  gemm(f128n, wf_sw, F(I_ENC + 10), kRows, 256, 128, f256, nullptr, 0, nullptr, 0, 0);
  max_over_k<<<dim3(kMG), dim3(256), 0, stream>>>(f256, kS, 256, fg256, nullptr, 0);
  enc_concat_k<<<dim3(kRows), dim3(128), 0, stream>>>(fg256, f256, cat512);
  zstat();
  gemm(cat512, wf_c1, F(I_ENC + 0), kRows, 512, 512, c1raw, nullptr, 0, stats, 1, 0);
  norm(c1raw, kRows, 512, stats, 1, 0, (float)kRows, F(I_ENC + 2), F(I_ENC + 1),
       1, nullptr, c1n, nullptr);
  gemm(c1n, wf_c2, F(I_ENC + 4), kRows, 256, 512, c2out, nullptr, 0, nullptr, 0, 0);
  max_over_k<<<dim3(kMG), dim3(256), 0, stream>>>(c2out, kS, 256, logits0, nullptr, 0);

  // ================= DGCNN (shared for dg1/dg2) =================
  auto run_dg = [&](int bi, const half_t* inF, int n5, half_t* out16, float* out32) {
    half_t* wtrans = convW(bi + 16, 128, 256);
    const int cin[4] = {128, 256, 512, 512};
    const int co[4]  = {256, 512, 512, 1024};
    const int wofs[4] = {bi + 2, bi + 5, bi + 8, bi + 11};
    const int fofs[4] = {0, 256, 768, 1280};
    gemm(inF, wtrans, F(bi + 15), kMG, 128, 256, ftr, nullptr, 0, nullptr, 0, 0);
    for (int i = 0; i < 4; ++i) {
      half_t* wl = convW(wofs[i], co[i], 2 * cin[i]);
      graph_feat_k<<<dim3(kMG), dim3(256), 0, stream>>>(ftr, idx16, cin[i], gf);
      zstat();
      gemm(gf, wl, nullptr, kMG * 16, co[i], 2 * cin[i], yraw, nullptr, 0, stats, 2,
           kGRP * 16);
      norm(yraw, (size_t)kMG * 16, co[i], stats, 2, kGRP * 16,
           (float)(kGRP * 16) * (co[i] / 4), F(wofs[i] - 1), F(wofs[i] - 2),
           2, nullptr, yn, nullptr);                       // lrelu(gn(.))
      max_over_k<<<dim3(kMG), dim3(256), 0, stream>>>(yn, 16, co[i], ftr,
                                                      feats + fofs[i], 2304);
    }
    half_t* w5 = convW(bi + 14, n5, 2304);
    zstat();
    gemm(feats, w5, nullptr, kMG, n5, 2304, l5raw, nullptr, 0, stats, 2, kGRP);
    norm(l5raw, kMG, n5, stats, 2, kGRP, (float)kGRP * (n5 / 4),
         F(bi + 13), F(bi + 12), 2, nullptr, out16, out32);
  };

  // dg1 -> logits (f32, straight into d_out)
  run_dg(I_DG1, logits0, kNT, nullptr, o_logits);

  // gumbel softmax + codebook lookup
  gumbel_softmax_k<<<dim3(kMG), dim3(256), 0, stream>>>(o_logits, F(I_NOISE), soft);
  gemm(soft, cbT, nullptr, kMG, 256, 8192, sampled, nullptr, 0, nullptr, 0, 0);

  // dg2 -> feature
  run_dg(I_DG2, sampled, 256, feature, nullptr);

  // ================= decoder =================
  gemm(feature, wm1, F(I_DEC + 10), kMG, 1024, 256, h1, nullptr, 1, nullptr, 0, 0);
  gemm(h1, wm2, F(I_DEC + 12), kMG, 1024, 1024, h2, nullptr, 1, nullptr, 0, 0);
  gemm(h2, wm3, F(I_DEC + 14), kMG, 96, 1024, nullptr, o_coarse, 0, nullptr, 0, 0);

  dec_feat_k<<<dim3(kRows), dim3(128), 0, stream>>>(feature, o_coarse, featA);
  zstat();
  gemm(featA, wfc1, F(I_DEC + 2), kRows, 512, 320, c1raw, nullptr, 0, stats, 1, 0);
  norm(c1raw, kRows, 512, stats, 1, 0, (float)kRows, F(I_DEC + 4), F(I_DEC + 3),
       1, nullptr, c1n, nullptr);
  zstat();
  gemm(c1n, wfc2, F(I_DEC + 6), kRows, 512, 512, cat512, nullptr, 0, stats, 1, 0);
  norm(cat512, kRows, 512, stats, 1, 0, (float)kRows, F(I_DEC + 8), F(I_DEC + 7),
       1, nullptr, fc2n, nullptr);

  gemm(fc2n, wxyz, F(I_DEC + 16), kRows, 3, 512, nullptr, xyzraw, 0, nullptr, 0, 0);
  fine_add_k<<<dim3((kRows * 3 + 255) / 256), dim3(256), 0, stream>>>(
      xyzraw, o_coarse, o_fine);
  gemm(fc2n, wemb, F(I_DEC + 0), kRows, 128, 512, nullptr, o_femb, 0, nullptr, 0, 0);
}
